// AttentionLayerImpressions_35828617183291
// MI455X (gfx1250) — matrix operand order repaired and checked by reading, not compiled, not run on hardware
//
#include <hip/hip_runtime.h>
#include <hip/hip_bf16.h>

#define T_DIM  512
#define B_DIM  64
#define H_DIM  1024
#define NI_DIM 1024

typedef __attribute__((ext_vector_type(16))) __bf16 v16bf;
typedef __attribute__((ext_vector_type(8)))  float  v8f;
typedef __attribute__((ext_vector_type(2)))  float  v2f;

// ---------------------------------------------------------------------------
// Kernel 1: pack W_h (fp32 row-major [K=H, N=H]) into bf16 B-fragment layout.
// Tile (kt, nt) covers K = kt*32..+31, N = nt*16..+15.
// Fragment element (lane l, elem e):  K = kt*32 + (l>>4)*8 + (e<8 ? e : e+8), N = nt*16 + (l&15)
// Stored contiguously: Whp[tile*512 + l*16 + e]  -> one 32B load/lane in GEMM.
// ---------------------------------------------------------------------------
__global__ void pack_wh_bf16(const float* __restrict__ Wh, __bf16* __restrict__ Whp) {
  int p    = blockIdx.x * blockDim.x + threadIdx.x;   // 0 .. H*H-1
  int e    = p & 15;
  int l    = (p >> 4) & 31;
  int tile = p >> 9;                                  // 0..2047
  int nt   = tile & 63;
  int kt   = tile >> 6;
  int k    = kt * 32 + ((l >> 4) << 3) + (e < 8 ? e : e + 8);
  int n    = nt * 16 + (l & 15);
  Whp[p] = (__bf16)Wh[k * H_DIM + n];
}

// ---------------------------------------------------------------------------
// Kernel 2: Ux = x_last[64,1024] @ W_x1[1024,1024], exact fp32 via WMMA 16x16x4.
// One wave per 16x16 output tile; 256 waves total.
// ---------------------------------------------------------------------------
__global__ void __launch_bounds__(256) ux_gemm(const float* __restrict__ X,
                                               const float* __restrict__ Wx,
                                               float* __restrict__ Ux) {
  const int tid = threadIdx.x;
#if __has_builtin(__builtin_amdgcn_wmma_f32_16x16x4_f32)
  const int lane = tid & 31;
  const int wv   = tid >> 5;
  const int tile = blockIdx.x * 8 + wv;     // 0..255
  const int m0   = (tile >> 6) * 16;        // 4 row tiles (B=64)
  const int n0   = (tile & 63) * 16;        // 64 col tiles (H=1024)
  const int ln   = lane & 15;
  const int lg   = lane >> 4;
  v8f c = {0.f, 0.f, 0.f, 0.f, 0.f, 0.f, 0.f, 0.f};
  for (int ks = 0; ks < NI_DIM / 4; ++ks) {
    const int kb = ks * 4;
    v2f a, b;
    // A 16x4 f32 layout: lane m = lane&15, elem e -> K = kb + lg*2 + e
    a.x = X[(m0 + ln) * NI_DIM + kb + lg * 2 + 0];
    a.y = X[(m0 + ln) * NI_DIM + kb + lg * 2 + 1];
    // B 4x16 f32 layout: lane n = lane&15, elem e -> K = kb + lg*2 + e
    b.x = Wx[(kb + lg * 2 + 0) * H_DIM + n0 + ln];
    b.y = Wx[(kb + lg * 2 + 1) * H_DIM + n0 + ln];
    c = __builtin_amdgcn_wmma_f32_16x16x4_f32(false, a, false, b, (short)0, c,
                                              false, false);
  }
#pragma unroll
  for (int v = 0; v < 8; ++v) {
    const int row = v + lg * 8;  // C layout: VGPR v, lanes 16-31 -> M = v+8
    Ux[(m0 + row) * H_DIM + n0 + ln] = c[v];
  }
#else
  // Scalar fallback (exact)
  const int gt = blockIdx.x * 256 + tid;  // 8192 threads, 8 outputs each
  for (int j = 0; j < 8; ++j) {
    int o = gt * 8 + j;
    int b = o >> 10;
    int h = o & (H_DIM - 1);
    float acc = 0.f;
    for (int k = 0; k < NI_DIM; ++k) acc += X[b * NI_DIM + k] * Wx[k * H_DIM + h];
    Ux[o] = acc;
  }
#endif
}

// ---------------------------------------------------------------------------
// Async global->LDS stage of one 32x32 fp32 A tile (4 KB), 16 B per thread.
// Raw row-major layout in LDS: tile[r][kk], r=0..31, kk=0..31.
// Tracked with ASYNCcnt; data never touches VGPRs.
// ---------------------------------------------------------------------------
__device__ __forceinline__ void issue_async_tile(const float* __restrict__ S,
                                                 float* ldsbuf, int m0, int ks,
                                                 int tid) {
  const int r = tid >> 3;                  // row 0..31
  const int q = tid & 7;                   // 16B chunk 0..7
  const float* gp = &S[(size_t)(m0 + r) * H_DIM + ks * 32 + (q << 2)];
  const unsigned loff = (unsigned)(uintptr_t)(&ldsbuf[r * 32 + (q << 2)]);
  asm volatile("global_load_async_to_lds_b128 %0, %1, off"
               :: "v"(loff), "v"(gp)
               : "memory");
}

// Build one 16x32 bf16 A fragment from the row-major fp32 LDS tile.
// v16bf elem e (lane group g) -> K = g*8 + (e<8 ? e : e+8).
__device__ __forceinline__ v16bf make_a_frag(const float* rowp) {
  const float4 fa = *(const float4*)(rowp +  0);   // K = g*8 + 0..3
  const float4 fb = *(const float4*)(rowp +  4);   // K = g*8 + 4..7
  const float4 fc = *(const float4*)(rowp + 16);   // K = g*8 + 16..19
  const float4 fd = *(const float4*)(rowp + 20);   // K = g*8 + 20..23
  v16bf a;
  a[0]  = (__bf16)fa.x;  a[1]  = (__bf16)fa.y;
  a[2]  = (__bf16)fa.z;  a[3]  = (__bf16)fa.w;
  a[4]  = (__bf16)fb.x;  a[5]  = (__bf16)fb.y;
  a[6]  = (__bf16)fb.z;  a[7]  = (__bf16)fb.w;
  a[8]  = (__bf16)fc.x;  a[9]  = (__bf16)fc.y;
  a[10] = (__bf16)fc.z;  a[11] = (__bf16)fc.w;
  a[12] = (__bf16)fd.x;  a[13] = (__bf16)fd.y;
  a[14] = (__bf16)fd.z;  a[15] = (__bf16)fd.w;
  return a;
}

// ---------------------------------------------------------------------------
// Kernel 3: scores. One workgroup per 32-row M-tile of flattened states_h
// [M=T*B=32768, K=H]. 8 waves x 128-col N-slices cover N=1024.
// Double-buffered async global->LDS A staging; bf16 WMMA 16x16x32, f32 acc;
// fused tanh(.+Ux)·v_a2 and per-row reduction.
// ---------------------------------------------------------------------------
__global__ void __launch_bounds__(256) scores_kernel(
    const float* __restrict__ S, const __bf16* __restrict__ Whp,
    const float* __restrict__ Ux, const float* __restrict__ va,
    float* __restrict__ e2) {
  __shared__ __align__(16) float ldsS[2][1024];  // 2 x 32x32 fp32 tiles (8 KB)
  __shared__ float e_part[32];

  const int tid  = threadIdx.x;
  const int lane = tid & 31;
  const int wv   = tid >> 5;          // 0..7
  const int m0   = blockIdx.x * 32;   // row block
  const int n0   = wv * 128;          // this wave's N slice
  const int ln   = lane & 15;
  const int lg   = lane >> 4;

  if (tid < 32) e_part[tid] = 0.f;

  const v8f vzero = {0.f, 0.f, 0.f, 0.f, 0.f, 0.f, 0.f, 0.f};
  v8f acc0[8], acc1[8];
#pragma unroll
  for (int nt = 0; nt < 8; ++nt) { acc0[nt] = vzero; acc1[nt] = vzero; }

  issue_async_tile(S, ldsS[0], m0, 0, tid);   // prime the pipeline

  for (int ks = 0; ks < H_DIM / 32; ++ks) {
    asm volatile("s_wait_asynccnt 0x0" ::: "memory");  // my share of tile ks
    __syncthreads();                                   // whole tile visible
    const float* cur = ldsS[ks & 1];
    if (ks + 1 < H_DIM / 32)                           // prefetch next tile
      issue_async_tile(S, ldsS[(ks + 1) & 1], m0, ks + 1, tid);

    const v16bf a0 = make_a_frag(cur + (ln)      * 32 + lg * 8);
    const v16bf a1 = make_a_frag(cur + (16 + ln) * 32 + lg * 8);
    const __bf16* bbase = Whp + ((size_t)ks * 64 + wv * 8) * 512;
#pragma unroll
    for (int nt = 0; nt < 8; ++nt) {
      const v16bf bf =
          *reinterpret_cast<const v16bf*>(bbase + (size_t)nt * 512 + (lane << 4));
      acc0[nt] = __builtin_amdgcn_wmma_f32_16x16x32_bf16(false, a0, false, bf,
                                                         (short)0, acc0[nt],
                                                         false, false);
      acc1[nt] = __builtin_amdgcn_wmma_f32_16x16x32_bf16(false, a1, false, bf,
                                                         (short)0, acc1[nt],
                                                         false, false);
    }
  }

  // Epilogue: e2 row partials = sum_n tanh(acc + Ux[b,n]) * v_a2[n]
#pragma unroll
  for (int mi = 0; mi < 2; ++mi) {
#pragma unroll
    for (int nt = 0; nt < 8; ++nt) {
      const int ncol = n0 + nt * 16 + ln;
      const float vva = va[ncol];
      const v8f acc = mi ? acc1[nt] : acc0[nt];
#pragma unroll
      for (int v = 0; v < 8; ++v) {
        const int row = mi * 16 + v + lg * 8;           // C layout
        const int b   = (m0 + row) & (B_DIM - 1);       // b = m % B
        float val = tanhf(acc[v] + Ux[b * H_DIM + ncol]) * vva;
        val += __shfl_xor(val, 1, 32);
        val += __shfl_xor(val, 2, 32);
        val += __shfl_xor(val, 4, 32);
        val += __shfl_xor(val, 8, 32);  // lanes 0 & 16 hold row sums
        if (ln == 0) atomicAdd(&e_part[row], val);      // ds_add_f32
      }
    }
  }
  __syncthreads();
  if (tid < 32) e2[m0 + tid] = e_part[tid];  // e2 flat [T*B], m = t*B+b
}

// ---------------------------------------------------------------------------
// Kernel 4: softmax over T axis. One block per batch column b.
// ---------------------------------------------------------------------------
__global__ void __launch_bounds__(256) softmax_time(const float* __restrict__ e2,
                                                    float* __restrict__ a2) {
  __shared__ float red[256];
  const int b   = blockIdx.x;
  const int tid = threadIdx.x;
  float v0 = e2[tid * B_DIM + b];
  float v1 = e2[(tid + 256) * B_DIM + b];
  red[tid] = fmaxf(v0, v1);
  __syncthreads();
  for (int s = 128; s > 0; s >>= 1) {
    if (tid < s) red[tid] = fmaxf(red[tid], red[tid + s]);
    __syncthreads();
  }
  const float mx = red[0];
  __syncthreads();
  const float x0 = __expf(v0 - mx);
  const float x1 = __expf(v1 - mx);
  red[tid] = x0 + x1;
  __syncthreads();
  for (int s = 128; s > 0; s >>= 1) {
    if (tid < s) red[tid] += red[tid + s];
    __syncthreads();
  }
  const float inv = 1.f / red[0];
  a2[tid * B_DIM + b]         = x0 * inv;
  a2[(tid + 256) * B_DIM + b] = x1 * inv;
}

// ---------------------------------------------------------------------------
// Kernel 5: context c2[b,h] = sum_t a2[t,b] * states_h[t,b,h]. Pure bandwidth.
// ---------------------------------------------------------------------------
__global__ void __launch_bounds__(256) context_kernel(
    const float* __restrict__ S, const float* __restrict__ a2,
    float* __restrict__ out) {
  const int b = blockIdx.y;
  const int h = blockIdx.x * 256 + threadIdx.x;
  float c = 0.f;
#pragma unroll 4
  for (int t = 0; t < T_DIM; ++t) {
    c += a2[t * B_DIM + b] * S[((size_t)t * B_DIM + b) * H_DIM + h];
  }
  out[b * H_DIM + h] = c;
}

// ---------------------------------------------------------------------------
extern "C" void kernel_launch(void* const* d_in, const int* in_sizes, int n_in,
                              void* d_out, int out_size, void* d_ws,
                              size_t ws_size, hipStream_t stream) {
  (void)in_sizes; (void)n_in; (void)out_size; (void)ws_size;
  const float* x_last = (const float*)d_in[0];  // [B, NI]
  const float* S      = (const float*)d_in[1];  // [T, B, H]
  const float* W_h    = (const float*)d_in[2];  // [H, H]
  const float* W_x1   = (const float*)d_in[3];  // [NI, H]
  const float* v_a2   = (const float*)d_in[4];  // [H]
  float* out = (float*)d_out;                   // [B, H]

  char* ws = (char*)d_ws;
  float*  Ux  = (float*)(ws);                   // 256 KB
  float*  e2  = (float*)(ws + 262144);          // 128 KB
  float*  a2  = (float*)(ws + 393216);          // 128 KB
  __bf16* Whp = (__bf16*)(ws + 524288);         // 2 MB packed bf16 W_h

  pack_wh_bf16<<<(H_DIM * H_DIM) / 256, 256, 0, stream>>>(W_h, Whp);
  ux_gemm<<<32, 256, 0, stream>>>(x_last, W_x1, Ux);
  scores_kernel<<<(T_DIM * B_DIM) / 32, 256, 0, stream>>>(S, Whp, Ux, v_a2, e2);
  softmax_time<<<B_DIM, 256, 0, stream>>>(e2, a2);
  context_kernel<<<dim3(H_DIM / 256, B_DIM), 256, 0, stream>>>(S, a2, out);
}
